// Energy_Layer_43379169689812
// MI455X (gfx1250) — compile-verified, matrix operand order checked by reading, not run on hardware
//
#include <hip/hip_runtime.h>
#include <hip/hip_bf16.h>

// ---------------- problem constants ----------------
#define NN 50000
#define EE 500000
#define DD 128
#define NTILES_N (NN / 16)   // 3125
#define NTILES_E (EE / 16)   // 31250
#define TILE_LDA 136         // padded LDS row stride in halves

typedef __attribute__((ext_vector_type(16))) _Float16 v16h;
typedef __attribute__((ext_vector_type(8)))  float    v8f;
typedef __attribute__((ext_vector_type(2)))  _Float16 h2;
typedef __attribute__((ext_vector_type(4)))  float    f4;

// ---------------- WMMA fragment helpers ----------------
// A fragment (16x32 f16, MxK) per ISA 7.12.2: lane m=lane&15, hi lanes get K+8;
// VGPR v<4 -> K=2v, v>=4 -> K=16+2(v-4); pairs packed per dword.
__device__ __forceinline__ v16h a_frag(const _Float16* At, int lane, int kbase) {
    const int m = lane & 15;
    const int hioff = (lane >> 4) << 3;
    v16h a;
#pragma unroll
    for (int v = 0; v < 8; ++v) {
        const int kb = (v & 4) ? (16 + ((v & 3) << 1)) : (v << 1);
        const int k  = kbase + kb + hioff;
        const h2 pr = *(const h2*)&At[m * TILE_LDA + k];  // 4B-aligned ds_load
        a[2 * v]     = pr[0];
        a[2 * v + 1] = pr[1];
    }
    return a;
}

// Load the 4 K-step B fragments of one 128x128 weight matrix (fragment-native
// layout produced by convert_weights): contiguous 32B per lane.
__device__ __forceinline__ void load_wfrags(const _Float16* wmat, int wave, int lane,
                                            v16h out[4]) {
#pragma unroll
    for (int ks = 0; ks < 4; ++ks)
        out[ks] = *(const v16h*)(wmat + (((ks * 8 + wave) * 32 + lane) << 4));
}

// One 16x16 output tile, K=128 (4 chained WMMAs).
__device__ __forceinline__ v8f gemm_tile(const _Float16* At, const v16h Bf[4], int lane) {
    v8f acc = {};
#pragma unroll
    for (int ks = 0; ks < 4; ++ks) {
        v16h a = a_frag(At, lane, ks * 32);
        acc = __builtin_amdgcn_wmma_f32_16x16x32_f16(false, a, false, Bf[ks],
                                                     (short)0, acc, false, false);
    }
    return acc;
}

// Scatter a v8f D tile (M = r + 8*hi, N = lane&15) into an LDS f16 activation tile.
__device__ __forceinline__ void store_act(_Float16* At, int lane, int colBase, v8f v) {
    const int n = lane & 15, hi = lane >> 4;
#pragma unroll
    for (int r = 0; r < 8; ++r)
        At[(r + hi * 8) * TILE_LDA + colBase + n] = (_Float16)v[r];
}

// Async 16B global -> LDS copy (CDNA5, tracked by ASYNCcnt).
__device__ __forceinline__ void async_copy_b128(void* lds_dst, const void* gsrc) {
    const unsigned lds_addr = (unsigned)(size_t)lds_dst;      // LDS byte offset
    const unsigned long long gaddr = (unsigned long long)(size_t)gsrc;
    asm volatile("global_load_async_to_lds_b128 %0, %1, off"
                 :: "v"(lds_addr), "v"(gaddr) : "memory");
}
__device__ __forceinline__ void wait_async0() {
    asm volatile("s_wait_asynccnt 0x0" ::: "memory");
}

// ---------------- kernel 0: weight transpose+convert to fragment layout ----------------
// Per matrix m: frag element flat index r = ((ks*8+col)*32+lane)*16+h, value W[o*128+k]
__global__ __launch_bounds__(256) void convert_weights(
    const float* w0, const float* w1, const float* w2,
    const float* w3, const float* w4, const float* w5,
    const float* w6, const float* w7, const float* w8,
    _Float16* wfrag) {
    const int t = blockIdx.x * 256 + threadIdx.x;
    if (t >= 9 * 16384) return;
    const float* Ws[9] = {w0, w1, w2, w3, w4, w5, w6, w7, w8};
    const int m = t >> 14;
    const int r = t & 16383;
    const int h    = r & 15;
    const int lane = (r >> 4) & 31;
    const int col  = (r >> 9) & 7;
    const int ks   = (r >> 12) & 3;
    const int p = h & 1, v = h >> 1;
    const int kb = (v & 4) ? (16 + ((v & 3) << 1)) : (v << 1);
    const int k  = ks * 32 + kb + p + ((lane >> 4) << 3);
    const int o  = col * 16 + (lane & 15);
    wfrag[t] = (_Float16)Ws[m][o * DD + k];
}

// ---------------- kernel 1: fused node encoders (h1 f32, p1/p2 f16) ----------------
__global__ __launch_bounds__(256) void encode_nodes(
    const float* __restrict__ x, const _Float16* __restrict__ wf,
    const float* __restrict__ bK, const float* __restrict__ bP1,
    const float* __restrict__ bP2,
    float* __restrict__ h1, _Float16* __restrict__ p1h, _Float16* __restrict__ p2h) {
    __shared__ _Float16 Atile[16 * TILE_LDA];
    const int tid = threadIdx.x, lane = tid & 31, wave = tid >> 5;
    const int colBase = wave * 16;
    const int n = lane & 15, hi = lane >> 4;

    v16h BK[4], BP1[4], BP2[4];
    load_wfrags(wf + 0 * 16384, wave, lane, BK);
    load_wfrags(wf + 1 * 16384, wave, lane, BP1);
    load_wfrags(wf + 2 * 16384, wave, lane, BP2);
    const float bbK  = bK[colBase + n];
    const float bbP1 = bP1[colBase + n];
    const float bbP2 = bP2[colBase + n];

    for (int tile = blockIdx.x; tile < NTILES_N; tile += gridDim.x) {
        const int rowBase = tile * 16;
        __syncthreads();
        for (int i = tid; i < 16 * DD; i += 256) {
            const int r = i >> 7, c = i & 127;
            Atile[r * TILE_LDA + c] = (_Float16)x[(size_t)(rowBase + r) * DD + c];
        }
        __syncthreads();

        v8f aK = {}, aP1 = {}, aP2 = {};
#pragma unroll
        for (int ks = 0; ks < 4; ++ks) {
            v16h a = a_frag(Atile, lane, ks * 32);
            aK  = __builtin_amdgcn_wmma_f32_16x16x32_f16(false, a, false, BK[ks],  (short)0, aK,  false, false);
            aP1 = __builtin_amdgcn_wmma_f32_16x16x32_f16(false, a, false, BP1[ks], (short)0, aP1, false, false);
            aP2 = __builtin_amdgcn_wmma_f32_16x16x32_f16(false, a, false, BP2[ks], (short)0, aP2, false, false);
        }
#pragma unroll
        for (int r = 0; r < 8; ++r) {
            const size_t idx = (size_t)(rowBase + r + hi * 8) * DD + colBase + n;
            h1[idx]  = aK[r] + bbK;
            p1h[idx] = (_Float16)(aP1[r] + bbP1);
            p2h[idx] = (_Float16)(aP2[r] + bbP2);
        }
    }
}

// ---------------- kernel 2: edge scatter-add of h1 into h1_agg ----------------
__global__ __launch_bounds__(256) void scatter_h1(
    const int* __restrict__ src, const int* __restrict__ dst,
    const float* __restrict__ h1, float* __restrict__ agg) {
    const int t = blockIdx.x * 256 + threadIdx.x;
    const int e = t >> 5, lane = t & 31;
    if (e >= EE) return;
    const int s = src[e], d = dst[e];
    const f4 v = *(const f4*)&h1[(size_t)s * DD + lane * 4];
    float* p = &agg[(size_t)d * DD + lane * 4];
    unsafeAtomicAdd(p + 0, v[0]);
    unsafeAtomicAdd(p + 1, v[1]);
    unsafeAtomicAdd(p + 2, v[2]);
    unsafeAtomicAdd(p + 3, v[3]);
}

// ---------------- kernel 3: Knode = MLP3(h1_agg) ----------------
__global__ __launch_bounds__(256) void knode_mlp(
    const float* __restrict__ agg, const _Float16* __restrict__ wf,  // K0..K2 frags
    const float* __restrict__ b0, const float* __restrict__ b1,
    const float* __restrict__ b2, float* __restrict__ Knode) {
    __shared__ _Float16 Abuf0[16 * TILE_LDA];
    __shared__ _Float16 Abuf1[16 * TILE_LDA];
    const int tid = threadIdx.x, lane = tid & 31, wave = tid >> 5;
    const int colBase = wave * 16;
    const int n = lane & 15, hi = lane >> 4;

    v16h W0[4], W1[4], W2[4];
    load_wfrags(wf + 0 * 16384, wave, lane, W0);
    load_wfrags(wf + 1 * 16384, wave, lane, W1);
    load_wfrags(wf + 2 * 16384, wave, lane, W2);
    const float bb0 = b0[colBase + n], bb1 = b1[colBase + n], bb2 = b2[colBase + n];

    for (int tile = blockIdx.x; tile < NTILES_N; tile += gridDim.x) {
        const int rowBase = tile * 16;
        __syncthreads();
        for (int i = tid; i < 16 * DD; i += 256) {
            const int r = i >> 7, c = i & 127;
            Abuf0[r * TILE_LDA + c] = (_Float16)agg[(size_t)(rowBase + r) * DD + c];
        }
        __syncthreads();

        v8f a0 = gemm_tile(Abuf0, W0, lane);          // layer0 + tanh
#pragma unroll
        for (int r = 0; r < 8; ++r) a0[r] = tanhf(a0[r] + bb0);
        store_act(Abuf1, lane, colBase, a0);
        __syncthreads();

        v8f a1 = gemm_tile(Abuf1, W1, lane);          // layer1 + relu
#pragma unroll
        for (int r = 0; r < 8; ++r) a1[r] = fmaxf(a1[r] + bb1, 0.0f);
        store_act(Abuf0, lane, colBase, a1);
        __syncthreads();

        v8f a2 = gemm_tile(Abuf0, W2, lane);          // layer2
#pragma unroll
        for (int r = 0; r < 8; ++r)
            Knode[(size_t)(rowBase + r + hi * 8) * DD + colBase + n] = a2[r] + bb2;
    }
}

// ---------------- kernel 4: edge MLP + Knode dot + global reduce ----------------
__global__ __launch_bounds__(256) void edge_mlp_reduce(
    const int* __restrict__ src, const int* __restrict__ dst,
    const _Float16* __restrict__ wf,  // U0..U2 frags
    const _Float16* __restrict__ p1h, const _Float16* __restrict__ p2h,
    const float* __restrict__ Knode,
    const float* __restrict__ b0, const float* __restrict__ b1,
    const float* __restrict__ b2, float* __restrict__ out) {
    __shared__ _Float16 Abuf0[16 * TILE_LDA];
    __shared__ _Float16 Abuf1[16 * TILE_LDA];
    __shared__ float Kn[16][DD];
    __shared__ int sIdx[16], dIdx[16];
    __shared__ float red[256];
    const int tid = threadIdx.x, lane = tid & 31, wave = tid >> 5;
    const int colBase = wave * 16;
    const int n = lane & 15, hi = lane >> 4;

    v16h W0[4], W1[4], W2[4];
    load_wfrags(wf + 0 * 16384, wave, lane, W0);
    load_wfrags(wf + 1 * 16384, wave, lane, W1);
    load_wfrags(wf + 2 * 16384, wave, lane, W2);
    const float bb0 = b0[colBase + n], bb1 = b1[colBase + n], bb2 = b2[colBase + n];

    float partial = 0.0f;
    for (int tile = blockIdx.x; tile < NTILES_E; tile += gridDim.x) {
        __syncthreads();
        if (tid < 16) {
            sIdx[tid] = src[tile * 16 + tid];
            dIdx[tid] = dst[tile * 16 + tid];
        }
        // prefetch next tile's index cachelines (global_prefetch_b8)
        const int nextTile = tile + gridDim.x;
        if (nextTile < NTILES_E && tid == 32) {
            __builtin_prefetch(&src[nextTile * 16], 0, 0);
            __builtin_prefetch(&dst[nextTile * 16], 0, 0);
        }
        __syncthreads();

        // async-gather Knode[src] tile (pure copy): global -> LDS, ASYNCcnt path
        for (int i = tid; i < 16 * 32; i += 256) {
            const int r = i >> 5, c4 = i & 31;
            async_copy_b128(&Kn[r][c4 * 4],
                            &Knode[(size_t)sIdx[r] * DD + c4 * 4]);
        }
        // gather A = p1[src] + p2[dst] (f16, packed adds) through VGPRs
        for (int i = tid; i < 16 * 64; i += 256) {
            const int r = i >> 6, c2 = i & 63;
            const h2 a = *(const h2*)&p1h[(size_t)sIdx[r] * DD + c2 * 2];
            const h2 b = *(const h2*)&p2h[(size_t)dIdx[r] * DD + c2 * 2];
            *(h2*)&Abuf0[r * TILE_LDA + c2 * 2] = a + b;
        }
        wait_async0();
        __syncthreads();

        v8f a0 = gemm_tile(Abuf0, W0, lane);
#pragma unroll
        for (int r = 0; r < 8; ++r) a0[r] = tanhf(a0[r] + bb0);
        store_act(Abuf1, lane, colBase, a0);
        __syncthreads();

        v8f a1 = gemm_tile(Abuf1, W1, lane);
#pragma unroll
        for (int r = 0; r < 8; ++r) a1[r] = fmaxf(a1[r] + bb1, 0.0f);
        store_act(Abuf0, lane, colBase, a1);
        __syncthreads();

        v8f a2 = gemm_tile(Abuf0, W2, lane);
#pragma unroll
        for (int r = 0; r < 8; ++r)
            partial += (a2[r] + bb2) * Kn[r + hi * 8][colBase + n];
    }

    red[tid] = partial;
    __syncthreads();
#pragma unroll
    for (int s = 128; s > 0; s >>= 1) {
        if (tid < s) red[tid] += red[tid + s];
        __syncthreads();
    }
    if (tid == 0) unsafeAtomicAdd(out, red[0]);
}

// ---------------- host launcher ----------------
extern "C" void kernel_launch(void* const* d_in, const int* in_sizes, int n_in,
                              void* d_out, int out_size, void* d_ws, size_t ws_size,
                              hipStream_t stream) {
    const float* x    = (const float*)d_in[0];
    const int*   src  = (const int*)d_in[1];
    const int*   dst  = (const int*)d_in[2];
    const float* WencK  = (const float*)d_in[3];  const float* bencK  = (const float*)d_in[4];
    const float* WencP1 = (const float*)d_in[5];  const float* bencP1 = (const float*)d_in[6];
    const float* WencP2 = (const float*)d_in[7];  const float* bencP2 = (const float*)d_in[8];
    const float* K0W = (const float*)d_in[9];   const float* K0b = (const float*)d_in[10];
    const float* K1W = (const float*)d_in[11];  const float* K1b = (const float*)d_in[12];
    const float* K2W = (const float*)d_in[13];  const float* K2b = (const float*)d_in[14];
    const float* U0W = (const float*)d_in[15];  const float* U0b = (const float*)d_in[16];
    const float* U1W = (const float*)d_in[17];  const float* U1b = (const float*)d_in[18];
    const float* U2W = (const float*)d_in[19];  const float* U2b = (const float*)d_in[20];

    // workspace carve-up
    char* ws = (char*)d_ws;
    _Float16* wfrag = (_Float16*)ws;                       size_t off = 9 * 16384 * 2;
    float*    h1    = (float*)(ws + off);                  off += (size_t)NN * DD * 4;
    float*    agg   = (float*)(ws + off);                  off += (size_t)NN * DD * 4;
    float*    Knode = (float*)(ws + off);                  off += (size_t)NN * DD * 4;
    _Float16* p1h   = (_Float16*)(ws + off);               off += (size_t)NN * DD * 2;
    _Float16* p2h   = (_Float16*)(ws + off);               off += (size_t)NN * DD * 2;

    hipMemsetAsync(agg, 0, (size_t)NN * DD * 4, stream);
    hipMemsetAsync(d_out, 0, sizeof(float), stream);

    convert_weights<<<(9 * 16384 + 255) / 256, 256, 0, stream>>>(
        WencK, WencP1, WencP2, K0W, K1W, K2W, U0W, U1W, U2W, wfrag);

    encode_nodes<<<512, 256, 0, stream>>>(x, wfrag, bencK, bencP1, bencP2, h1, p1h, p2h);

    scatter_h1<<<(EE * 32) / 256, 256, 0, stream>>>(src, dst, h1, agg);

    knode_mlp<<<512, 256, 0, stream>>>(agg, wfrag + 3 * 16384, K0b, K1b, K2b, Knode);

    edge_mlp_reduce<<<1024, 256, 0, stream>>>(src, dst, wfrag + 6 * 16384,
                                              p1h, p2h, Knode, U0b, U1b, U2b,
                                              (float*)d_out);
}
